// AttentionHead_33646773797609
// MI455X (gfx1250) — compile-verified, matrix operand order checked by reading, not run on hardware
//
#include <hip/hip_runtime.h>
#include <hip/hip_bf16.h>
#include <stdint.h>

// ---------------------------------------------------------------------------
// AttentionHead for MI455X (gfx1250): bf16 WMMA pipeline, fp32 accumulate.
// B=4, S=2048, DK=1024.  All intermediates (~166MB) fit in the 192MB L2.
// Per-wave GEMM tile: 64x64 -> 16 WMMAs per k-step vs 8 fragment loads
// (1.0 global_load_b128 per WMMA).
// ---------------------------------------------------------------------------

#define SEQ   2048
#define DKDIM 1024
#define NBAT  4

typedef __attribute__((ext_vector_type(16))) __bf16 bf16x16;
typedef __attribute__((ext_vector_type(8)))  float  f32x8;

__device__ __forceinline__ uint16_t f32_to_bf16_rne(float f) {
  uint32_t u = __builtin_bit_cast(uint32_t, f);
  u += 0x7FFFu + ((u >> 16) & 1u);
  return (uint16_t)(u >> 16);
}

// Fragment loader for v_wmma_f32_16x16x32_bf16.
// A fragment (16x32, row r = lane%16) per ISA: halves 0-7 = K k0+8*hi..+7,
// halves 8-15 = K k0+16+8*hi..+7  (hi = lane/16).  The B fragment (32x16,
// B[k][n]) uses the identical per-lane pattern with n = lane%16, so one loader
// serves both whenever the "row" direction of the source matrix runs over K.
// base must point at the first row of the 16-row group; row length == K.
__device__ __forceinline__ bf16x16 frag_load(const uint16_t* __restrict__ base,
                                             int K, int k0) {
  const int lane = threadIdx.x & 31;
  const int r    = lane & 15;
  const int hi   = lane >> 4;
  const uint16_t* p = base + (size_t)r * K + k0 + 8 * hi;
  union { uint4 u[2]; bf16x16 v; } tmp;
  tmp.u[0] = *reinterpret_cast<const uint4*>(p);
  tmp.u[1] = *reinterpret_cast<const uint4*>(p + 16);
  return tmp.v;
}

#define WMMA_BF16(A, B, C) \
  __builtin_amdgcn_wmma_f32_16x16x32_bf16(false, (A), false, (B), (short)0, (C), false, false)

// ---------------------------------------------------------------------------
// f32 -> bf16 cast, 4 elements/thread
// ---------------------------------------------------------------------------
__global__ void __launch_bounds__(256)
cast_f32_bf16(const float4* __restrict__ src, ushort4* __restrict__ dst, int n4) {
  int i = blockIdx.x * 256 + threadIdx.x;
  if (i < n4) {
    float4 v = src[i];
    ushort4 o;
    o.x = f32_to_bf16_rne(v.x);
    o.y = f32_to_bf16_rne(v.y);
    o.z = f32_to_bf16_rne(v.z);
    o.w = f32_to_bf16_rne(v.w);
    dst[i] = o;
  }
}

// ---------------------------------------------------------------------------
// Projection GEMM: Y[m,n] = sum_k X[m,k] * W[n,k]   (y = x @ W^T)
// X: [M,K] bf16 row-major,  W: [N,K] bf16 row-major, Y: bf16.
// transposed==1 stores per-batch Y^T: Y[b][n][s]  (used for V so that the
// final attn@V GEMM can load its B fragments as contiguous rows).
// Block: 256 threads = 8 waves; wave tile 64x64; block tile 512x64.
// ---------------------------------------------------------------------------
__global__ void __launch_bounds__(256)
gemm_proj_bf16(const uint16_t* __restrict__ X,
               const uint16_t* __restrict__ W,
               uint16_t* __restrict__ Y,
               int N, int K, int transposed, int seq) {
  const int wave = threadIdx.x >> 5;
  const int m0   = (blockIdx.y * 8 + wave) * 64;
  const int n0   = blockIdx.x * 64;

  f32x8 acc[4][4] = {};
  const uint16_t* Arow[4];
#pragma unroll
  for (int u = 0; u < 4; ++u) Arow[u] = X + (size_t)(m0 + 16 * u) * K;

  for (int k0 = 0; k0 < K; k0 += 32) {
    bf16x16 a0 = frag_load(Arow[0], K, k0);
    bf16x16 a1 = frag_load(Arow[1], K, k0);
    bf16x16 a2 = frag_load(Arow[2], K, k0);
    bf16x16 a3 = frag_load(Arow[3], K, k0);
#pragma unroll
    for (int t = 0; t < 4; ++t) {
      bf16x16 b = frag_load(W + (size_t)(n0 + 16 * t) * K, K, k0);
      acc[0][t] = WMMA_BF16(a0, b, acc[0][t]);
      acc[1][t] = WMMA_BF16(a1, b, acc[1][t]);
      acc[2][t] = WMMA_BF16(a2, b, acc[2][t]);
      acc[3][t] = WMMA_BF16(a3, b, acc[3][t]);
    }
  }

  const int lane = threadIdx.x & 31;
  const int r    = lane & 15;
  const int hi   = lane >> 4;
#pragma unroll
  for (int u = 0; u < 4; ++u) {
#pragma unroll
    for (int t = 0; t < 4; ++t) {
#pragma unroll
      for (int f = 0; f < 8; ++f) {
        const int m = m0 + 16 * u + 8 * hi + f;
        const int n = n0 + 16 * t + r;
        const uint16_t h = f32_to_bf16_rne(acc[u][t][f]);
        if (!transposed) {
          Y[(size_t)m * N + n] = h;
        } else {
          const int b = m / seq, s = m - b * seq;    // tiles never cross batches
          Y[((size_t)b * N + n) * seq + s] = h;
        }
      }
    }
  }
}

// ---------------------------------------------------------------------------
// Batched GEMM, bf16 in / fp32 out:  C[m,n] = scale * sum_k A[m,k] * Bsrc[n,k]
// Used for (a) scores: A=qp, Bsrc=kp rows (S = qp @ kp^T / 32)
//          (b) output: A=attn, Bsrc=vp^T rows (O = attn @ vp)
// In both uses the row length of Bsrc equals K.
// ---------------------------------------------------------------------------
__global__ void __launch_bounds__(256)
gemm_bf16_f32(const uint16_t* __restrict__ A, size_t strideA,
              const uint16_t* __restrict__ Bsrc, size_t strideB,
              float* __restrict__ C, size_t strideC,
              int N, int K, float scale) {
  const int zb = blockIdx.z;
  const uint16_t* Ab = A    + (size_t)zb * strideA;
  const uint16_t* Bb = Bsrc + (size_t)zb * strideB;
  float*          Cb = C    + (size_t)zb * strideC;

  const int wave = threadIdx.x >> 5;
  const int m0   = (blockIdx.y * 8 + wave) * 64;
  const int n0   = blockIdx.x * 64;

  f32x8 acc[4][4] = {};
  const uint16_t* Arow[4];
#pragma unroll
  for (int u = 0; u < 4; ++u) Arow[u] = Ab + (size_t)(m0 + 16 * u) * K;

  for (int k0 = 0; k0 < K; k0 += 32) {
    bf16x16 a0 = frag_load(Arow[0], K, k0);
    bf16x16 a1 = frag_load(Arow[1], K, k0);
    bf16x16 a2 = frag_load(Arow[2], K, k0);
    bf16x16 a3 = frag_load(Arow[3], K, k0);
#pragma unroll
    for (int t = 0; t < 4; ++t) {
      bf16x16 b = frag_load(Bb + (size_t)(n0 + 16 * t) * K, K, k0);
      acc[0][t] = WMMA_BF16(a0, b, acc[0][t]);
      acc[1][t] = WMMA_BF16(a1, b, acc[1][t]);
      acc[2][t] = WMMA_BF16(a2, b, acc[2][t]);
      acc[3][t] = WMMA_BF16(a3, b, acc[3][t]);
    }
  }

  const int lane = threadIdx.x & 31;
  const int r    = lane & 15;
  const int hi   = lane >> 4;
#pragma unroll
  for (int u = 0; u < 4; ++u) {
#pragma unroll
    for (int t = 0; t < 4; ++t) {
#pragma unroll
      for (int f = 0; f < 8; ++f) {
        const int m = m0 + 16 * u + 8 * hi + f;
        const int n = n0 + 16 * t + r;
        Cb[(size_t)m * N + n] = acc[u][t][f] * scale;
      }
    }
  }
}

// ---------------------------------------------------------------------------
// Row softmax: P[row, :] = exp(S - max) / sum(exp(S - max)), bf16 out.
// Mathematically identical to the reference's exp(s)/sum(exp(s)).
// One 256-thread block per row (n = 2048).
// ---------------------------------------------------------------------------
__global__ void __launch_bounds__(256)
softmax_rows(const float* __restrict__ S, uint16_t* __restrict__ P, int n) {
  __shared__ float red[256];
  const size_t row = blockIdx.x;
  const float* s = S + row * (size_t)n;
  uint16_t*    p = P + row * (size_t)n;

  float m = -3.4e38f;
  for (int i = threadIdx.x; i < n; i += 256) m = fmaxf(m, s[i]);
  red[threadIdx.x] = m;
  __syncthreads();
  for (int w = 128; w > 0; w >>= 1) {
    if ((int)threadIdx.x < w) red[threadIdx.x] = fmaxf(red[threadIdx.x], red[threadIdx.x + w]);
    __syncthreads();
  }
  m = red[0];
  __syncthreads();

  float sum = 0.f;
  for (int i = threadIdx.x; i < n; i += 256) sum += __expf(s[i] - m);
  red[threadIdx.x] = sum;
  __syncthreads();
  for (int w = 128; w > 0; w >>= 1) {
    if ((int)threadIdx.x < w) red[threadIdx.x] += red[threadIdx.x + w];
    __syncthreads();
  }
  const float inv = 1.f / red[0];

  for (int i = threadIdx.x; i < n; i += 256)
    p[i] = f32_to_bf16_rne(__expf(s[i] - m) * inv);
}

// ---------------------------------------------------------------------------
// Host-side orchestration
// ---------------------------------------------------------------------------
extern "C" void kernel_launch(void* const* d_in, const int* in_sizes, int n_in,
                              void* d_out, int out_size, void* d_ws, size_t ws_size,
                              hipStream_t stream) {
  (void)in_sizes; (void)n_in; (void)out_size; (void)ws_size;

  const float* q  = (const float*)d_in[0];
  const float* k  = (const float*)d_in[1];
  const float* v  = (const float*)d_in[2];
  const float* wq = (const float*)d_in[3];
  const float* wk = (const float*)d_in[4];
  const float* wv = (const float*)d_in[5];
  float* out = (float*)d_out;

  const int M  = NBAT * SEQ;               // 8192
  const size_t xElems = (size_t)M * DKDIM; // 8.39M
  const size_t wElems = (size_t)DKDIM * DKDIM;
  const size_t sElems = (size_t)NBAT * SEQ * SEQ;

  // Workspace layout (256B aligned)
  uint8_t* ws = (uint8_t*)d_ws;
  size_t off = 0;
  auto alloc = [&](size_t bytes) -> uint8_t* {
    uint8_t* p = ws + off;
    off = (off + bytes + 255) & ~(size_t)255;
    return p;
  };
  uint16_t* q_bf  = (uint16_t*)alloc(xElems * 2);  // attn (bf16, 32MB) aliases
  uint16_t* k_bf  = (uint16_t*)alloc(xElems * 2);  //   q_bf..k_bf after use
  uint16_t* v_bf  = (uint16_t*)alloc(xElems * 2);
  uint16_t* wq_bf = (uint16_t*)alloc(wElems * 2);
  uint16_t* wk_bf = (uint16_t*)alloc(wElems * 2);
  uint16_t* wv_bf = (uint16_t*)alloc(wElems * 2);
  uint16_t* qp    = (uint16_t*)alloc(xElems * 2);
  uint16_t* kp    = (uint16_t*)alloc(xElems * 2);
  uint16_t* vpT   = (uint16_t*)alloc(xElems * 2);  // [B][DK][S]
  float*    sc    = (float*)   alloc(sElems * 4);  // fp32 scores
  uint16_t* attn  = q_bf;                          // alias: bf16 probs

  // 1) casts
  {
    const int n4x = (int)(xElems / 4), n4w = (int)(wElems / 4);
    dim3 bx((n4x + 255) / 256), bw((n4w + 255) / 256);
    cast_f32_bf16<<<bx, 256, 0, stream>>>((const float4*)q,  (ushort4*)q_bf,  n4x);
    cast_f32_bf16<<<bx, 256, 0, stream>>>((const float4*)k,  (ushort4*)k_bf,  n4x);
    cast_f32_bf16<<<bx, 256, 0, stream>>>((const float4*)v,  (ushort4*)v_bf,  n4x);
    cast_f32_bf16<<<bw, 256, 0, stream>>>((const float4*)wq, (ushort4*)wq_bf, n4w);
    cast_f32_bf16<<<bw, 256, 0, stream>>>((const float4*)wk, (ushort4*)wk_bf, n4w);
    cast_f32_bf16<<<bw, 256, 0, stream>>>((const float4*)wv, (ushort4*)wv_bf, n4w);
  }

  // 2) projections: qp = q@wq^T, kp = k@wk^T, vpT = (v@wv^T)^T per batch
  {
    dim3 grid(DKDIM / 64, M / 512);
    gemm_proj_bf16<<<grid, 256, 0, stream>>>(q_bf, wq_bf, qp,  DKDIM, DKDIM, 0, SEQ);
    gemm_proj_bf16<<<grid, 256, 0, stream>>>(k_bf, wk_bf, kp,  DKDIM, DKDIM, 0, SEQ);
    gemm_proj_bf16<<<grid, 256, 0, stream>>>(v_bf, wv_bf, vpT, DKDIM, DKDIM, 1, SEQ);
  }

  // 3) scores: sc = qp @ kp^T / sqrt(DK)
  {
    dim3 grid(SEQ / 64, SEQ / 512, NBAT);
    gemm_bf16_f32<<<grid, 256, 0, stream>>>(
        qp, (size_t)SEQ * DKDIM, kp, (size_t)SEQ * DKDIM,
        sc, (size_t)SEQ * SEQ, SEQ, DKDIM, 1.0f / 32.0f);
  }

  // 4) softmax rows -> bf16 probabilities
  softmax_rows<<<dim3(NBAT * SEQ), 256, 0, stream>>>(sc, attn, SEQ);

  // 5) out = attn @ vp   (B fragments read rows of vpT)
  {
    dim3 grid(DKDIM / 64, SEQ / 512, NBAT);
    gemm_bf16_f32<<<grid, 256, 0, stream>>>(
        attn, (size_t)SEQ * SEQ, vpT, (size_t)DKDIM * SEQ,
        out, (size_t)SEQ * DKDIM, DKDIM, SEQ, 1.0f);
  }
}
